// ReinforceLayer_28054726378324
// MI455X (gfx1250) — matrix operand order repaired and checked by reading, not compiled, not run on hardware
//
#include <hip/hip_runtime.h>

// ---------------------------------------------------------------------------
// ReinforceLayer pipeline for MI455X (gfx1250, wave32, WMMA).
// Implicit-GEMM conv/fc via v_wmma_f32_16x16x32_f16, f16 activations.
// Fully templated shapes -> constant-offset VMEM, LDS double buffering,
// tap-major im2col K ordering (each 32-wide K chunk sits in one 3x3 tap).
// ---------------------------------------------------------------------------

typedef __attribute__((ext_vector_type(16))) _Float16 v16h;
typedef __attribute__((ext_vector_type(8)))  float    v8f;

#define NB   64
#define CIc  3
#define HIc  96
#define WIc  96
#define GGc  8
#define HGc  16
#define WGc  16
#define HIDc 18432
#define NLIN 512

struct __align__(16) h8 { _Float16 v[8]; };

// A-fragment (16x32 f16): lanes<16 hold K {0..7,16..23}, lanes>=16 hold K {8..15,24..31}
__device__ __forceinline__ v16h load_frag_a(const _Float16* row, int lane) {
  const int b0 = (lane >> 4) * 8;
  union { v16h v; h8 h[2]; } u;
  u.h[0] = *reinterpret_cast<const h8*>(row + b0);
  u.h[1] = *reinterpret_cast<const h8*>(row + b0 + 16);
  return u.v;
}
// B-fragment (32x16 f16): lanes<16 hold K {0..7,16..23}, lanes>=16 hold K {8..15,24..31}
__device__ __forceinline__ v16h load_frag_b(const _Float16* row, int lane) {
  const int b0 = (lane >> 4) * 8;
  union { v16h v; h8 h[2]; } u;
  u.h[0] = *reinterpret_cast<const h8*>(row + b0);
  u.h[1] = *reinterpret_cast<const h8*>(row + b0 + 16);
  return u.v;
}

// ---------------------------------------------------------------------------
// Fast implicit-GEMM 3x3 conv (pad=1) + bias + ReLU, CIN % 32 == 0.
// K ordering tap-major: k = tap*CIN + ci. Block: 128 thr -> 32(M) x 128(N).
// Double-buffered LDS; global loads of chunk c+1 overlap WMMA of chunk c.
// ---------------------------------------------------------------------------
template<int CIN, int H, int Wd, int COUT>
__global__ __launch_bounds__(128)
void conv3x3_wmma_fast(const _Float16* __restrict__ X, const float* __restrict__ Wt,
                       const float* __restrict__ bias, _Float16* __restrict__ Y)
{
  constexpr int K  = CIN * 9;
  constexpr int HW = H * Wd;
  constexpr int NCHUNK = K / 32;
  const int nBase = blockIdx.x * 128;
  const int mBase = blockIdx.y * 32;
  const int tid  = threadIdx.x;
  const int lane = tid & 31;
  const int wv   = tid >> 5;
  const int waveM = wv >> 1, waveN = wv & 1;

  __shared__ __align__(16) _Float16 As[2][32][40];   // 2 x 32(M) x 32(K)
  __shared__ __align__(16) _Float16 Bs[2][128][40];  // 2 x 128(N) x 32(K)

  // this thread's im2col column
  const int n  = nBase + tid;
  const int bb = n / HW;
  const int rm = n - bb * HW;
  const int hh = rm / Wd;
  const int ww = rm - hh * Wd;
  const _Float16* baseX = X + (size_t)bb * CIN * HW;
  const int am  = tid >> 2;
  const int aks = (tid & 3) * 8;

  float    aw[8];
  _Float16 bx[32];
  bool     bvalid;

  // ---- load phase: chunk k0 -> registers (constant-offset gathers) ----
  auto load_chunk = [&](int k0) {
    const int tap = k0 / CIN;                 // scalar, 0..8
    const int ciB = k0 - tap * CIN;
    const int t3  = tap / 3;
    const int dh  = t3 - 1;
    const int dw  = (tap - t3 * 3) - 1;
    const float* wp = Wt + ((mBase + am) * CIN + ciB + aks) * 9 + tap;
    #pragma unroll
    for (int i = 0; i < 8; ++i) aw[i] = wp[i * 9];
    const int h2 = hh + dh, w2 = ww + dw;
    bvalid = ((unsigned)h2 < (unsigned)H) & ((unsigned)w2 < (unsigned)Wd);
    const int hs = min(max(h2, 0), H - 1);
    const int wsafe = min(max(w2, 0), Wd - 1);
    const _Float16* px = baseX + ciB * HW + hs * Wd + wsafe;
    #pragma unroll
    for (int i = 0; i < 32; ++i) bx[i] = px[i * HW];   // constant offsets
  };
  // ---- store phase: registers -> LDS (b128 stores) ----
  auto store_chunk = [&](int bi) {
    h8 ha;
    #pragma unroll
    for (int i = 0; i < 8; ++i) ha.v[i] = (_Float16)aw[i];
    *reinterpret_cast<h8*>(&As[bi][am][aks]) = ha;
    #pragma unroll
    for (int g = 0; g < 4; ++g) {
      h8 hb;
      #pragma unroll
      for (int i = 0; i < 8; ++i) hb.v[i] = bvalid ? bx[g * 8 + i] : (_Float16)0.f;
      *reinterpret_cast<h8*>(&Bs[bi][tid][g * 8]) = hb;
    }
  };

  v8f acc[4] = {v8f{}, v8f{}, v8f{}, v8f{}};
  auto compute = [&](int bi) {
    const v16h a = load_frag_a(&As[bi][waveM * 16 + (lane & 15)][0], lane);
    #pragma unroll
    for (int f = 0; f < 4; ++f) {
      const v16h b = load_frag_b(&Bs[bi][waveN * 64 + f * 16 + (lane & 15)][0], lane);
      acc[f] = __builtin_amdgcn_wmma_f32_16x16x32_f16(
          false, a, false, b, (short)0, acc[f], false, false);
    }
  };

  load_chunk(0); store_chunk(0);
  __syncthreads();
  for (int c = 0; c < NCHUNK - 1; ++c) {
    load_chunk((c + 1) * 32);      // global loads in flight during WMMA
    compute(c & 1);
    store_chunk((c + 1) & 1);
    __syncthreads();
  }
  compute((NCHUNK - 1) & 1);

  // ---- epilogue: bias + ReLU, constant-offset NCHW stores ----
  const int mrow = waveM * 16 + 8 * (lane >> 4);
  const int co0  = mBase + mrow;
  const float4 bv0 = *reinterpret_cast<const float4*>(bias + co0);
  const float4 bv1 = *reinterpret_cast<const float4*>(bias + co0 + 4);
  const float bb8[8] = {bv0.x, bv0.y, bv0.z, bv0.w, bv1.x, bv1.y, bv1.z, bv1.w};
  #pragma unroll
  for (int f = 0; f < 4; ++f) {
    const int p  = nBase + waveN * 64 + f * 16 + (lane & 15);
    const int b2 = p / HW;                       // constant divide
    const int r2 = p - b2 * HW;                  // == h2*Wd + w2
    _Float16* yp = Y + (size_t)(b2 * COUT + co0) * HW + r2;
    #pragma unroll
    for (int r = 0; r < 8; ++r)
      yp[r * HW] = (_Float16)fmaxf(acc[f][r] + bb8[r], 0.f);
  }
}

// ---------------------------------------------------------------------------
// conv1a: Cin=3, K=27 (pad to 32) -- single K chunk, everything constant.
// ---------------------------------------------------------------------------
__global__ __launch_bounds__(128)
void conv3x3_c3_wmma(const _Float16* __restrict__ X, const float* __restrict__ Wt,
                     const float* __restrict__ bias, _Float16* __restrict__ Y)
{
  constexpr int K = 27, H = HIc, Wd = WIc, Cout = 32;
  constexpr int HW = H * Wd;
  const int nBase = blockIdx.x * 128;
  const int tid  = threadIdx.x;
  const int lane = tid & 31;
  const int wv   = tid >> 5;
  const int waveM = wv >> 1, waveN = wv & 1;

  __shared__ __align__(16) _Float16 As[32][40];
  __shared__ __align__(16) _Float16 Bs[128][40];

  const int n  = nBase + tid;
  const int bb = n / HW;
  const int rm = n - bb * HW;
  const int hh = rm / Wd;
  const int ww = rm - hh * Wd;
  const _Float16* baseX = X + (size_t)bb * 3 * HW;

  {
    const int m  = tid >> 2;
    const int ks = (tid & 3) * 8;
    h8 hv;
    #pragma unroll
    for (int i = 0; i < 8; ++i) {
      const int k  = ks + i;
      const int kc = (k < K) ? k : (K - 1);
      const float v = Wt[m * K + kc];
      hv.v[i] = (k < K) ? (_Float16)v : (_Float16)0.f;
    }
    *reinterpret_cast<h8*>(&As[m][ks]) = hv;
  }
  #pragma unroll
  for (int g = 0; g < 4; ++g) {
    h8 hv;
    #pragma unroll
    for (int i = 0; i < 8; ++i) {
      const int k  = g * 8 + i;
      const int kc = (k < K) ? k : (K - 1);
      const int ci = kc / 9;
      const int tap = kc - ci * 9;
      const int h2 = hh + tap / 3 - 1;
      const int w2 = ww + tap % 3 - 1;
      const bool valid = (k < K) & ((unsigned)h2 < (unsigned)H) & ((unsigned)w2 < (unsigned)Wd);
      const int hs = min(max(h2, 0), H - 1);
      const int wsafe = min(max(w2, 0), Wd - 1);
      const _Float16 v = baseX[(ci * H + hs) * Wd + wsafe];
      hv.v[i] = valid ? v : (_Float16)0.f;
    }
    *reinterpret_cast<h8*>(&Bs[tid][g * 8]) = hv;
  }
  __syncthreads();

  v8f acc[4] = {v8f{}, v8f{}, v8f{}, v8f{}};
  const v16h a = load_frag_a(&As[waveM * 16 + (lane & 15)][0], lane);
  #pragma unroll
  for (int f = 0; f < 4; ++f) {
    const v16h b = load_frag_b(&Bs[waveN * 64 + f * 16 + (lane & 15)][0], lane);
    acc[f] = __builtin_amdgcn_wmma_f32_16x16x32_f16(
        false, a, false, b, (short)0, acc[f], false, false);
  }

  const int mrow = waveM * 16 + 8 * (lane >> 4);
  const float4 bv0 = *reinterpret_cast<const float4*>(bias + mrow);
  const float4 bv1 = *reinterpret_cast<const float4*>(bias + mrow + 4);
  const float bb8[8] = {bv0.x, bv0.y, bv0.z, bv0.w, bv1.x, bv1.y, bv1.z, bv1.w};
  #pragma unroll
  for (int f = 0; f < 4; ++f) {
    const int p  = nBase + waveN * 64 + f * 16 + (lane & 15);
    const int b2 = p / HW;
    const int r2 = p - b2 * HW;
    _Float16* yp = Y + (size_t)(b2 * Cout + mrow) * HW + r2;
    #pragma unroll
    for (int r = 0; r < 8; ++r)
      yp[r * HW] = (_Float16)fmaxf(acc[f][r] + bb8[r], 0.f);
  }
}

// ---------------------------------------------------------------------------
// Split-K FC GEMM: partial[s][64][512] = X[64, k-slice] @ W[k-slice, 512].
// Templated K / slice -> constant-offset staging; double-buffered LDS.
// ---------------------------------------------------------------------------
template<int K, int KSLICE>
__global__ __launch_bounds__(128)
void fc_wmma_splitk(const _Float16* __restrict__ X, const float* __restrict__ Wt,
                    float* __restrict__ partial)
{
  constexpr int M = 64, N = NLIN;
  constexpr int NCHUNK = KSLICE / 32;
  const int nBase = blockIdx.x * 128;
  const int mBase = blockIdx.y * 32;
  const int slice = blockIdx.z;
  const int kBeg  = slice * KSLICE;
  const int tid  = threadIdx.x;
  const int lane = tid & 31;
  const int wv   = tid >> 5;
  const int waveM = wv >> 1, waveN = wv & 1;

  __shared__ __align__(16) _Float16 As[2][32][40];
  __shared__ __align__(16) _Float16 Bs[2][128][40];

  const int am  = tid >> 2;
  const int aks = (tid & 3) * 8;
  const _Float16* xp  = X + (size_t)(mBase + am) * K + kBeg + aks;
  const float*    wpB = Wt + (size_t)kBeg * N + nBase + tid;

  h8    ax;
  float bxf[32];

  auto load_chunk = [&](int c) {
    ax = *reinterpret_cast<const h8*>(xp + c * 32);
    const float* wp = wpB + (size_t)c * 32 * N;
    #pragma unroll
    for (int i = 0; i < 32; ++i) bxf[i] = wp[i * N];   // constant offsets
  };
  auto store_chunk = [&](int bi) {
    *reinterpret_cast<h8*>(&As[bi][am][aks]) = ax;
    #pragma unroll
    for (int g = 0; g < 4; ++g) {
      h8 hb;
      #pragma unroll
      for (int i = 0; i < 8; ++i) hb.v[i] = (_Float16)bxf[g * 8 + i];
      *reinterpret_cast<h8*>(&Bs[bi][tid][g * 8]) = hb;
    }
  };

  v8f acc[4] = {v8f{}, v8f{}, v8f{}, v8f{}};
  auto compute = [&](int bi) {
    const v16h a = load_frag_a(&As[bi][waveM * 16 + (lane & 15)][0], lane);
    #pragma unroll
    for (int f = 0; f < 4; ++f) {
      const v16h b = load_frag_b(&Bs[bi][waveN * 64 + f * 16 + (lane & 15)][0], lane);
      acc[f] = __builtin_amdgcn_wmma_f32_16x16x32_f16(
          false, a, false, b, (short)0, acc[f], false, false);
    }
  };

  load_chunk(0); store_chunk(0);
  __syncthreads();
  for (int c = 0; c < NCHUNK - 1; ++c) {
    load_chunk(c + 1);
    compute(c & 1);
    store_chunk((c + 1) & 1);
    __syncthreads();
  }
  compute((NCHUNK - 1) & 1);

  float* dst = partial + (size_t)slice * M * N;
  const int mrow = waveM * 16 + 8 * (lane >> 4);
  #pragma unroll
  for (int f = 0; f < 4; ++f) {
    const int nn = nBase + waveN * 64 + f * 16 + (lane & 15);
    float* dp = dst + (size_t)(mBase + mrow) * N + nn;
    #pragma unroll
    for (int r = 0; r < 8; ++r) dp[r * N] = acc[f][r];
  }
}

// deterministic split-K reduce + bias + ReLU + f16 convert
__global__ __launch_bounds__(256)
void fc_reduce(const float* __restrict__ partial, const float* __restrict__ bias,
               _Float16* __restrict__ Y, int MN, int N, int S)
{
  const int idx = blockIdx.x * 256 + threadIdx.x;
  if (idx >= MN) return;
  float s = bias[idx % N];
  const float* p = partial + idx;
  for (int sl = 0; sl < S; ++sl) { s += *p; p += MN; }
  Y[idx] = (_Float16)fmaxf(s, 0.f);
}

// ---------------------------------------------------------------------------
// Elementwise helpers
// ---------------------------------------------------------------------------
__global__ __launch_bounds__(256)
void f32_to_f16(const float* __restrict__ src, _Float16* __restrict__ dst, int n) {
  const int i = blockIdx.x * 256 + threadIdx.x;
  if (i < n) dst[i] = (_Float16)src[i];
}

__global__ __launch_bounds__(256)
void maxpool_f16(const _Float16* __restrict__ X, _Float16* __restrict__ Y,
                 int Bn, int C, int H, int Wd, int k)
{
  const int Ho = H / k, Wo = Wd / k;
  const int idx = blockIdx.x * 256 + threadIdx.x;
  const int total = Bn * C * Ho * Wo;
  if (idx >= total) return;
  const int wo = idx % Wo;
  const int ho = (idx / Wo) % Ho;
  const int c  = (idx / (Wo * Ho)) % C;
  const int b  = idx / (Wo * Ho * C);
  float m = -__builtin_inff();
  for (int dy = 0; dy < k; ++dy)
    for (int dx = 0; dx < k; ++dx) {
      float v = (float)X[((b * C + c) * H + ho * k + dy) * Wd + wo * k + dx];
      m = fmaxf(m, v);
    }
  Y[idx] = (_Float16)m;
}

// fc3: prep[b][j] = x2[b,:] . wl3[:,j] + bl3[j]   (64 x 24, tiny)
__global__ __launch_bounds__(32)
void fc3_prep(const _Float16* __restrict__ X, const float* __restrict__ Wt,
              const float* __restrict__ bias, float* __restrict__ prep)
{
  const int b = blockIdx.x, j = threadIdx.x;
  if (j < GGc * 3) {
    float s = bias[j];
    for (int k = 0; k < NLIN; ++k)
      s += (float)X[b * NLIN + k] * Wt[k * (GGc * 3) + j];
    prep[b * (GGc * 3) + j] = s;
  }
}

// softplus/sigmoid head + 16x16 patch extraction; writes result/means/sigmas
__global__ __launch_bounds__(128)
void head_extract(const float* __restrict__ prep, const float* __restrict__ noise,
                  const float* __restrict__ image, float* __restrict__ out)
{
  const int bg = blockIdx.x;
  const int b  = bg / GGc;
  const int g  = bg - b * GGc;
  __shared__ int pt[2];

  if (threadIdx.x < 2) {
    const int i = threadIdx.x;
    const float mean = prep[b * (GGc * 3) + g * 3 + i];
    const float sraw = prep[b * (GGc * 3) + g * 3 + 2];
    const float x    = sraw + 2.0f;                       // SIGMA_BOOST
    const float sp   = (x > 20.f) ? x : log1pf(expf(x));  // softplus
    const float sigma = (sp + 1e-7f) * (float)(HIc - 1);  // * 95
    const float sample = mean + sigma * noise[bg * 2 + i];
    const float s = 1.f / (1.f + expf(-sample));
    int p = (int)rintf(s * (float)(HIc - HGc - 1));       // * 79, round-half-even
    p = min(max(p, 0), HIc - HGc);                        // dynamic_slice clamp
    pt[i] = p;
    out[NB * GGc * CIc * HGc * WGc + bg * 2 + i] = mean;            // means
    out[NB * GGc * CIc * HGc * WGc + NB * GGc * 2 + bg * 2 + i] = sigma; // sigmas
  }
  __syncthreads();

  for (int idx = threadIdx.x; idx < CIc * HGc * WGc; idx += 128) {
    const int c  = idx / (HGc * WGc);
    const int r  = idx - c * HGc * WGc;
    const int hh = r / WGc;
    const int wcol = r - hh * WGc;
    out[bg * (CIc * HGc * WGc) + idx] =
        image[((b * CIc + c) * HIc + pt[0] + hh) * WIc + pt[1] + wcol];
  }
}

// ---------------------------------------------------------------------------
// Launcher
// ---------------------------------------------------------------------------
extern "C" void kernel_launch(void* const* d_in, const int* in_sizes, int n_in,
                              void* d_out, int out_size, void* d_ws, size_t ws_size,
                              hipStream_t stream) {
  (void)in_sizes; (void)n_in; (void)out_size; (void)ws_size;
  const float* image = (const float*)d_in[0];
  const float* noise = (const float*)d_in[1];
  const float* w1a = (const float*)d_in[2];  const float* b1a = (const float*)d_in[3];
  const float* w1b = (const float*)d_in[4];  const float* b1b = (const float*)d_in[5];
  const float* w2a = (const float*)d_in[6];  const float* b2a = (const float*)d_in[7];
  const float* w2b = (const float*)d_in[8];  const float* b2b = (const float*)d_in[9];
  const float* w3a = (const float*)d_in[10]; const float* b3a = (const float*)d_in[11];
  const float* w3b = (const float*)d_in[12]; const float* b3b = (const float*)d_in[13];
  const float* wl1 = (const float*)d_in[14]; const float* bl1 = (const float*)d_in[15];
  const float* wl2 = (const float*)d_in[16]; const float* bl2 = (const float*)d_in[17];
  const float* wl3 = (const float*)d_in[18]; const float* bl3 = (const float*)d_in[19];
  float* out = (float*)d_out;

  char* ws = (char*)d_ws;
  const size_t BUFBYTES = (size_t)NB * 32 * HIc * WIc * sizeof(_Float16); // 36 MB
  _Float16* bufA = (_Float16*)ws;
  _Float16* bufB = (_Float16*)(ws + BUFBYTES);
  float*    P    = (float*)(ws + 2 * BUFBYTES);   // split-K partials (<= 3 MB)

  // image fp32 -> f16
  { const int n = NB * CIc * HIc * WIc;
    f32_to_f16<<<n / 256, 256, 0, stream>>>(image, bufA, n); }

  // conv stack (N = B*H*W always divisible by 128)
  conv3x3_c3_wmma<<<dim3(NB*HIc*WIc/128, 1), 128, 0, stream>>>(bufA, w1a, b1a, bufB);
  conv3x3_wmma_fast<32, 96, 96, 32 ><<<dim3(NB*96*96/128, 1), 128, 0, stream>>>(bufB, w1b, b1b, bufA);
  maxpool_f16<<<NB*32*24*24/256, 256, 0, stream>>>(bufA, bufB, NB, 32, 96, 96, 4);
  conv3x3_wmma_fast<32, 24, 24, 64 ><<<dim3(NB*24*24/128, 2), 128, 0, stream>>>(bufB, w2a, b2a, bufA);
  conv3x3_wmma_fast<64, 24, 24, 64 ><<<dim3(NB*24*24/128, 2), 128, 0, stream>>>(bufA, w2b, b2b, bufB);
  maxpool_f16<<<NB*64*12*12/256, 256, 0, stream>>>(bufB, bufA, NB, 64, 24, 24, 2);
  conv3x3_wmma_fast<64, 12, 12, 128><<<dim3(NB*12*12/128, 4), 128, 0, stream>>>(bufA, w3a, b3a, bufB);
  conv3x3_wmma_fast<128,12, 12, 128><<<dim3(NB*12*12/128, 4), 128, 0, stream>>>(bufB, w3b, b3b, bufA);

  // MLP: (64,18432)->(64,512)->(64,512), split-K for occupancy
  fc_wmma_splitk<HIDc, 768><<<dim3(NLIN/128, 2, 24), 128, 0, stream>>>(bufA, wl1, P);
  fc_reduce<<<(64*NLIN)/256, 256, 0, stream>>>(P, bl1, bufB, 64*NLIN, NLIN, 24);
  fc_wmma_splitk<NLIN, 128><<<dim3(NLIN/128, 2, 4), 128, 0, stream>>>(bufB, wl2, P);
  fc_reduce<<<(64*NLIN)/256, 256, 0, stream>>>(P, bl2, bufA, 64*NLIN, NLIN, 4);

  // head
  float* prep = (float*)P;
  fc3_prep<<<NB, 32, 0, stream>>>(bufA, wl3, bl3, prep);
  head_extract<<<NB * GGc, 128, 0, stream>>>(prep, noise, image, out);
}
